// MultiheadAttention_23441931502128
// MI455X (gfx1250) — matrix operand order repaired and checked by reading, not compiled, not run on hardware
//
#include <hip/hip_runtime.h>
#include <hip/hip_bf16.h>

#define BATCH  2
#define FEAT   1024
#define SEQL   2048
#define NHEAD  16
#define DH     64
#define KSZ    7
#define KBLK   64

typedef __attribute__((ext_vector_type(16))) _Float16 v16h;
typedef __attribute__((ext_vector_type(8)))  float    v8f;
typedef __attribute__((ext_vector_type(4)))  unsigned v4u;
typedef __attribute__((ext_vector_type(4)))  int      v4i;
typedef __attribute__((ext_vector_type(8)))  int      v8i;

#if __has_builtin(__builtin_amdgcn_tensor_load_to_lds) && \
    __has_builtin(__builtin_amdgcn_s_wait_tensorcnt)
#define USE_TDM 1
#define NBUF 2
#else
#define USE_TDM 0
#define NBUF 1
#endif

#if USE_TDM
// Issue a 2D tensor-tile DMA into LDS (Tensor Data Mover, ISA ch.8).
// Units are 4-byte elements (data_size=2). LDS rows get `pada+1` dwords of
// padding after every 2^(padi+1) dwords, matching our padded LDS tiles.
// Wave-scalar: call from one wave only; EXEC is ignored by TDM.
static __device__ inline void tdm_load_2d(const void* gsrc, unsigned ldsoff,
                                          unsigned td0, unsigned td1,
                                          unsigned stride0,
                                          unsigned tile0, unsigned tile1,
                                          unsigned padi, unsigned pada) {
  unsigned long long ga = (unsigned long long)(size_t)gsrc;
  v4u g0 = { 1u,                                   // count=1, user descriptor
             ldsoff,                               // lds_addr [63:32]
             (unsigned)ga,                         // global_addr low
             (unsigned)((ga >> 32) & 0x01FFFFFFull) | (2u << 30) }; // type=2
  v8i g1 = { (int)((2u << 16) | (1u << 20) | (padi << 22) | (pada << 25)),
             (int)((td0 & 0xFFFFu) << 16),         // tensor_dim0 [79:48]
             (int)((td0 >> 16) | ((td1 & 0xFFFFu) << 16)),
             (int)((td1 >> 16) | (tile0 << 16)),   // tile_dim0 [127:112]
             (int)tile1,                           // tile_dim1 (tile_dim2=0)
             (int)stride0,                         // tensor_dim0_stride low
             0, 0 };
  v4i z4 = {0, 0, 0, 0};
#if __clang_major__ >= 23
  v8i z8 = {0, 0, 0, 0, 0, 0, 0, 0};
  __builtin_amdgcn_tensor_load_to_lds(g0, g1, z4, z4, z8, 0);
#else
  __builtin_amdgcn_tensor_load_to_lds(g0, g1, z4, z4, 0);
#endif
}
#endif

// 16 halves (32B) from LDS via two b128 loads (row strides are 16B-multiples).
static __device__ inline v16h lds_ld16h(const _Float16* p) {
  union { uint4 u[2]; v16h h; } t;
  t.u[0] = *(const uint4*)(p);
  t.u[1] = *(const uint4*)(p + 8);
  return t.h;
}

// ---------------------------------------------------------------- cast f32->f16
__global__ __launch_bounds__(256) void cast_f32_to_f16(const float* __restrict__ s,
                                                       _Float16* __restrict__ d, int n) {
  int i = (blockIdx.x * 256 + threadIdx.x) * 4;
  if (i + 3 < n) {
    float4 v = *(const float4*)(s + i);
    union { uint2 u; _Float16 h[4]; } t;
    t.h[0] = (_Float16)v.x; t.h[1] = (_Float16)v.y;
    t.h[2] = (_Float16)v.z; t.h[3] = (_Float16)v.w;
    *(uint2*)(d + i) = t.u;
  }
}

// --------------------------------------------- q/k depthwise conv + RoPE + scale
__global__ __launch_bounds__(256) void qkconv_rope_kernel(
    const float* __restrict__ x, const float* __restrict__ qw,
    const float* __restrict__ kw,
    _Float16* __restrict__ qbuf, _Float16* __restrict__ kbuf) {
  int tid = blockIdx.x * 256 + threadIdx.x;     // BATCH * 512 pairs * SEQL
  int w  = tid & (SEQL - 1);
  int cp = (tid >> 11) & 511;
  int b  = tid >> 20;
  int c0 = cp * 2;
  const float* s0 = x + ((size_t)b * FEAT + c0) * SEQL;
  const float* s1 = s0 + SEQL;
  float q0 = 0.f, q1 = 0.f, k0 = 0.f, k1 = 0.f;
  #pragma unroll
  for (int t = 0; t < KSZ; ++t) {
    int ww = w + t - KSZ / 2;
    if (ww >= 0 && ww < SEQL) {
      float a0 = s0[ww], a1 = s1[ww];
      q0 += a0 * qw[c0 * KSZ + t];       q1 += a1 * qw[(c0 + 1) * KSZ + t];
      k0 += a0 * kw[c0 * KSZ + t];       k1 += a1 * kw[(c0 + 1) * KSZ + t];
    }
  }
  int head = c0 >> 6;
  int din  = c0 & 63;
  int pair = din >> 1;
  float inv_freq = __powf(10000.0f, -(float)(2 * pair) * (1.0f / (float)DH));
  float ang = (float)w * inv_freq;
  float sn, cs;
  __sincosf(ang, &sn, &cs);
  float qe = q0 * cs - q1 * sn, qo = q1 * cs + q0 * sn;
  float ke = k0 * cs - k1 * sn, ko = k1 * cs + k0 * sn;
  const float scale = 0.03125f;                  // 1/sqrt(FEAT), folded into Q
  qe *= scale; qo *= scale;
  size_t idx = (((size_t)b * NHEAD + head) * SEQL + w) * DH + din;
  union { unsigned int u; _Float16 h[2]; } pq, pk;
  pq.h[0] = (_Float16)qe; pq.h[1] = (_Float16)qo;
  pk.h[0] = (_Float16)ke; pk.h[1] = (_Float16)ko;
  *(unsigned int*)(qbuf + idx) = pq.u;
  *(unsigned int*)(kbuf + idx) = pk.u;
}

// ------------------------------------------------ v depthwise conv (post-GEMM)
__global__ __launch_bounds__(256) void vconv_kernel(
    const float* __restrict__ v0, const float* __restrict__ cw,
    _Float16* __restrict__ vbuf) {
  int tid = blockIdx.x * 256 + threadIdx.x;
  int w  = tid & (SEQL - 1);
  int cp = (tid >> 11) & 511;
  int b  = tid >> 20;
  int c0 = cp * 2;
  const float* s0 = v0 + ((size_t)b * FEAT + c0) * SEQL;
  const float* s1 = s0 + SEQL;
  float a0 = 0.f, a1 = 0.f;
  #pragma unroll
  for (int t = 0; t < KSZ; ++t) {
    int ww = w + t - KSZ / 2;
    if (ww >= 0 && ww < SEQL) {
      a0 += s0[ww] * cw[c0 * KSZ + t];
      a1 += s1[ww] * cw[(c0 + 1) * KSZ + t];
    }
  }
  int head = c0 >> 6, din = c0 & 63;
  size_t idx = (((size_t)b * NHEAD + head) * SEQL + w) * DH + din;
  union { unsigned int u; _Float16 h[2]; } t2;
  t2.h[0] = (_Float16)a0; t2.h[1] = (_Float16)a1;
  *(unsigned int*)(vbuf + idx) = t2.u;
}

// --------------------------------------------------------- v_proj GEMM (WMMA)
// v0[b][o][w] = sum_h W[o][h] * X[b][h][w]; 128x128 C block per workgroup.
// Double-buffered TDM pipeline: DMA of tile it+1 overlaps WMMA on tile it.
__global__ __launch_bounds__(256) void gemm_vproj_kernel(
    const _Float16* __restrict__ W,   // [FEAT][FEAT]
    const _Float16* __restrict__ X,   // [BATCH][FEAT][SEQL]
    float* __restrict__ Cout) {       // [BATCH][FEAT][SEQL]
  __shared__ __attribute__((aligned(128))) _Float16 at[NBUF][128][40];
  __shared__ __attribute__((aligned(128))) _Float16 bt[NBUF][32][136];
  const int lane = threadIdx.x & 31;
  const int wave = threadIdx.x >> 5;
  const int w0 = blockIdx.x * 128;
  const int o0 = blockIdx.y * 128;
  const int b  = blockIdx.z;
  const _Float16* Xb = X + (size_t)b * FEAT * SEQL;

  v8f cf[8];
  #pragma unroll
  for (int i = 0; i < 8; ++i)
    #pragma unroll
    for (int r = 0; r < 8; ++r) cf[i][r] = 0.f;

  const int NIT = FEAT / 32;
#if USE_TDM
  if (wave == 0) {
    tdm_load_2d(W + (size_t)o0 * FEAT, (unsigned)(size_t)&at[0][0][0],
                512, 1024, 512, 16, 128, 3, 3);
    tdm_load_2d(Xb + w0, (unsigned)(size_t)&bt[0][0][0],
                1024, 1024, 1024, 64, 32, 5, 3);
    __builtin_amdgcn_s_wait_tensorcnt(0);
  }
  __syncthreads();
#else
  const int arow = threadIdx.x >> 1, aseg = (threadIdx.x & 1) * 16;
  const int brow = threadIdx.x >> 3, bseg = (threadIdx.x & 7) * 16;
#endif

  for (int it = 0; it < NIT; ++it) {
    const int k0 = it * 32;
#if USE_TDM
    const int cur = it & 1;
    if (wave == 0 && it + 1 < NIT) {   // prefetch next tiles via DMA
      tdm_load_2d(W + (size_t)o0 * FEAT + k0 + 32,
                  (unsigned)(size_t)&at[cur ^ 1][0][0],
                  512, 1024, 512, 16, 128, 3, 3);
      tdm_load_2d(Xb + (size_t)(k0 + 32) * SEQL + w0,
                  (unsigned)(size_t)&bt[cur ^ 1][0][0],
                  1024, 1024, 1024, 64, 32, 5, 3);
    }
#else
    const int cur = 0;
    __syncthreads();
    const _Float16* ga = W + (size_t)(o0 + arow) * FEAT + k0 + aseg;
    *(uint4*)&at[0][arow][aseg]     = *(const uint4*)ga;
    *(uint4*)&at[0][arow][aseg + 8] = *(const uint4*)(ga + 8);
    const _Float16* gb = Xb + (size_t)(k0 + brow) * SEQL + w0 + bseg;
    *(uint4*)&bt[0][brow][bseg]     = *(const uint4*)gb;
    *(uint4*)&bt[0][brow][bseg + 8] = *(const uint4*)(gb + 8);
    if (k0 + 32 < FEAT) {
      __builtin_prefetch(ga + 32, 0, 1);
      __builtin_prefetch(gb + (size_t)32 * SEQL, 0, 1);
    }
    __syncthreads();
#endif
    // A fragment: documented 16-bit A 16x32 lane layout (K pairs contiguous).
    v16h af;
    {
      int m = wave * 16 + (lane & 15);
      int hs = (lane >> 4) << 3;
      #pragma unroll
      for (int j = 0; j < 8; ++j) {
        int kb = ((j & 4) << 2) + hs + ((j & 3) << 1);
        ((unsigned int*)&af)[j] = *(const unsigned int*)&at[cur][m][kb];
      }
    }
    #pragma unroll
    for (int nt = 0; nt < 8; ++nt) {
      v16h bf;   // lane%16 = N col, 16 K rows (table order)
      #pragma unroll
      for (int e = 0; e < 16; ++e)
        bf[e] = bt[cur][((e & 8) << 1) + ((lane >> 4) << 3) + (e & 7)][nt * 16 + (lane & 15)];
      cf[nt] = __builtin_amdgcn_wmma_f32_16x16x32_f16(false, af, false, bf,
                                                      (short)0, cf[nt], false, false);
    }
#if USE_TDM
    if (wave == 0) __builtin_amdgcn_s_wait_tensorcnt(0);
    __syncthreads();                    // retire readers + publish next buffer
#endif
  }
  const int mrow = o0 + wave * 16 + 8 * (lane >> 4);
  const int ncol = lane & 15;
  float* outb = Cout + (size_t)b * FEAT * SEQL;
  #pragma unroll
  for (int nt = 0; nt < 8; ++nt)
    #pragma unroll
    for (int r = 0; r < 8; ++r)
      outb[(size_t)(mrow + r) * SEQL + w0 + nt * 16 + ncol] = cf[nt][r];
}

// ------------------------------------------------------ flash attention (WMMA)
// One block = (b, head, 128 q rows); 8 waves x 16 q rows each.
// Double-buffered TDM staging of K/V tiles overlaps the DMA with softmax/WMMA.
__global__ __launch_bounds__(256) void attn_kernel(
    const _Float16* __restrict__ qbuf, const _Float16* __restrict__ kbuf,
    const _Float16* __restrict__ vbuf, _Float16* __restrict__ abuf) {
  __shared__ __attribute__((aligned(128))) _Float16 kt[NBUF][KBLK][DH + 8];
  __shared__ __attribute__((aligned(128))) _Float16 vt[NBUF][KBLK][DH + 8];
  __shared__ _Float16 pt[8][16][KBLK + 8];                 // per-wave P [m][j]
  const int lane = threadIdx.x & 31;
  const int wave = threadIdx.x >> 5;
  const int head = blockIdx.y, b = blockIdx.z;
  const size_t hoff = ((size_t)b * NHEAD + head) * SEQL * DH;
  const _Float16* Q = qbuf + hoff;
  const _Float16* K = kbuf + hoff;
  const _Float16* V = vbuf + hoff;
  const int m0 = blockIdx.x * 128 + wave * 16;

  // Q A-fragments (d = 64 -> two 16x32 k-chunks), loaded once per wave.
  v16h aq[2];
  {
    const _Float16* qrow = Q + (size_t)(m0 + (lane & 15)) * DH;
    int hs = (lane >> 4) << 3;
    #pragma unroll
    for (int kc = 0; kc < 2; ++kc)
      #pragma unroll
      for (int j = 0; j < 8; ++j) {
        int kb = kc * 32 + ((j & 4) << 2) + hs + ((j & 3) << 1);
        ((unsigned int*)&aq[kc])[j] = *(const unsigned int*)(qrow + kb);
      }
  }

  v8f acc[4];
  #pragma unroll
  for (int i = 0; i < 4; ++i)
    #pragma unroll
    for (int r = 0; r < 8; ++r) acc[i][r] = 0.f;
  float mrow[8], lrow[8];
  #pragma unroll
  for (int r = 0; r < 8; ++r) { mrow[r] = -1e30f; lrow[r] = 0.f; }

  const int NIT = SEQL / KBLK;
#if USE_TDM
  if (wave == 0) {
    // 64 rows x 32 dwords, pad 4 dwords / 32 dwords -> 72-half rows.
    tdm_load_2d(K, (unsigned)(size_t)&kt[0][0][0], 32, SEQL, 32, 32, KBLK, 4, 3);
    tdm_load_2d(V, (unsigned)(size_t)&vt[0][0][0], 32, SEQL, 32, 32, KBLK, 4, 3);
    __builtin_amdgcn_s_wait_tensorcnt(0);
  }
  __syncthreads();
#else
  const int sj   = threadIdx.x >> 2;
  const int sseg = (threadIdx.x & 3) * 16;
#endif

  for (int it = 0; it < NIT; ++it) {
#if USE_TDM
    const int cur = it & 1;
    if (wave == 0 && it + 1 < NIT) {   // DMA next K/V block during compute
      tdm_load_2d(K + (size_t)(it + 1) * KBLK * DH,
                  (unsigned)(size_t)&kt[cur ^ 1][0][0], 32, SEQL, 32, 32, KBLK, 4, 3);
      tdm_load_2d(V + (size_t)(it + 1) * KBLK * DH,
                  (unsigned)(size_t)&vt[cur ^ 1][0][0], 32, SEQL, 32, 32, KBLK, 4, 3);
    }
#else
    const int cur = 0;
    const int kb0 = it * KBLK;
    __syncthreads();
    {
      const _Float16* krow = K + (size_t)(kb0 + sj) * DH + sseg;
      *(uint4*)&kt[0][sj][sseg]     = *(const uint4*)krow;
      *(uint4*)&kt[0][sj][sseg + 8] = *(const uint4*)(krow + 8);
      const _Float16* vrow = V + (size_t)(kb0 + sj) * DH + sseg;
      *(uint4*)&vt[0][sj][sseg]     = *(const uint4*)vrow;
      *(uint4*)&vt[0][sj][sseg + 8] = *(const uint4*)(vrow + 8);
      if (kb0 + KBLK < SEQL) {
        __builtin_prefetch(krow + (size_t)KBLK * DH, 0, 1);
        __builtin_prefetch(vrow + (size_t)KBLK * DH, 0, 1);
      }
    }
    __syncthreads();
#endif

    // S = Q K^T: 16 rows x 64 key cols (4 n-tiles).
    v8f sf[4];
    #pragma unroll
    for (int jt = 0; jt < 4; ++jt) {
      v8f c;
      #pragma unroll
      for (int r = 0; r < 8; ++r) c[r] = 0.f;
      #pragma unroll
      for (int kc = 0; kc < 2; ++kc) {
        // B fragment of K^T: lane%16 = j col, 16 d rows (table order).
        v16h bf;
        #pragma unroll
        for (int e = 0; e < 16; ++e) bf[e] = kt[cur][jt * 16 + (lane & 15)][kc * 32 + ((e & 8) << 1) + ((lane >> 4) << 3) + (e & 7)];
        c = __builtin_amdgcn_wmma_f32_16x16x32_f16(false, aq[kc], false, bf,
                                                   (short)0, c, false, false);
      }
      sf[jt] = c;
    }

    // Online softmax (rows live per-VGPR-slot across a 16-lane half).
    float fac[8];
    #pragma unroll
    for (int r = 0; r < 8; ++r) {
      float mx = fmaxf(fmaxf(sf[0][r], sf[1][r]), fmaxf(sf[2][r], sf[3][r]));
      #pragma unroll
      for (int off = 8; off > 0; off >>= 1) mx = fmaxf(mx, __shfl_xor(mx, off, 32));
      float mn = fmaxf(mrow[r], mx);
      fac[r] = __expf(mrow[r] - mn);
      mrow[r] = mn;
      float sum = 0.f;
      #pragma unroll
      for (int jt = 0; jt < 4; ++jt) {
        float p = __expf(sf[jt][r] - mn);
        sf[jt][r] = p;
        sum += p;
      }
      #pragma unroll
      for (int off = 8; off > 0; off >>= 1) sum += __shfl_xor(sum, off, 32);
      lrow[r] = lrow[r] * fac[r] + sum;
    }

    // P: C layout -> per-wave LDS tile -> A fragments (LDS in-order per wave).
    #pragma unroll
    for (int jt = 0; jt < 4; ++jt)
      #pragma unroll
      for (int r = 0; r < 8; ++r)
        pt[wave][r + 8 * (lane >> 4)][jt * 16 + (lane & 15)] = (_Float16)sf[jt][r];
    v16h ap[2];
    {
      int mr = lane & 15;
      int hs = (lane >> 4) << 3;
      #pragma unroll
      for (int kc = 0; kc < 2; ++kc)
        #pragma unroll
        for (int j = 0; j < 8; ++j) {
          int kbi = kc * 32 + ((j & 4) << 2) + hs + ((j & 3) << 1);
          ((unsigned int*)&ap[kc])[j] = *(const unsigned int*)&pt[wave][mr][kbi];
        }
    }

    // O = diag(fac)*O + P V
    #pragma unroll
    for (int dt = 0; dt < 4; ++dt) {
      #pragma unroll
      for (int r = 0; r < 8; ++r) acc[dt][r] *= fac[r];
      #pragma unroll
      for (int kc = 0; kc < 2; ++kc) {
        v16h bv;  // lane%16 = d col, 16 j rows (table order)
        #pragma unroll
        for (int e = 0; e < 16; ++e)
          bv[e] = vt[cur][kc * 32 + ((e & 8) << 1) + ((lane >> 4) << 3) + (e & 7)][dt * 16 + (lane & 15)];
        acc[dt] = __builtin_amdgcn_wmma_f32_16x16x32_f16(false, ap[kc], false, bv,
                                                         (short)0, acc[dt], false, false);
      }
    }
#if USE_TDM
    if (wave == 0) __builtin_amdgcn_s_wait_tensorcnt(0);
    __syncthreads();                    // retire readers + publish next buffer
#endif
  }

  // Normalize and store attention output as f16 [b][w][f].
  float invl[8];
  #pragma unroll
  for (int r = 0; r < 8; ++r) invl[r] = 1.0f / lrow[r];
  const int ncol = lane & 15, mh = 8 * (lane >> 4);
  #pragma unroll
  for (int dt = 0; dt < 4; ++dt)
    #pragma unroll
    for (int r = 0; r < 8; ++r) {
      float o = acc[dt][r] * invl[r];
      abuf[((size_t)b * SEQL + m0 + r + mh) * FEAT + head * DH + dt * 16 + ncol] = (_Float16)o;
    }
}

// -------------------------------------------------------- out_proj GEMM (WMMA)
// out[b][o][w] = sum_f O[o][f] * a[b][w][f]; A tile via TDM (double-buffered),
// transposed B tile staged manually in the same pipeline.
__global__ __launch_bounds__(256) void gemm_outproj_kernel(
    const _Float16* __restrict__ Wo,   // [FEAT][FEAT]
    const _Float16* __restrict__ A,    // [BATCH][SEQL][FEAT]
    float* __restrict__ Out) {         // [BATCH][FEAT][SEQL]
  __shared__ __attribute__((aligned(128))) _Float16 at[NBUF][128][40];
  __shared__ __attribute__((aligned(128))) _Float16 bt[NBUF][32][136];
  const int lane = threadIdx.x & 31;
  const int wave = threadIdx.x >> 5;
  const int w0 = blockIdx.x * 128;
  const int o0 = blockIdx.y * 128;
  const int b  = blockIdx.z;

  v8f cf[8];
  #pragma unroll
  for (int i = 0; i < 8; ++i)
    #pragma unroll
    for (int r = 0; r < 8; ++r) cf[i][r] = 0.f;

  const int bww = threadIdx.x & 127, bks = (threadIdx.x >> 7) * 16;
  const _Float16* Arow = A + ((size_t)b * SEQL + w0 + bww) * FEAT + bks;
  const int NIT = FEAT / 32;

#if USE_TDM
  {
    if (wave == 0)
      tdm_load_2d(Wo + (size_t)o0 * FEAT, (unsigned)(size_t)&at[0][0][0],
                  512, 1024, 512, 16, 128, 3, 3);
    union { uint4 u[2]; _Float16 h[16]; } tb;
    tb.u[0] = *(const uint4*)Arow;
    tb.u[1] = *(const uint4*)(Arow + 8);
    #pragma unroll
    for (int e = 0; e < 16; ++e) bt[0][bks + e][bww] = tb.h[e];
    if (wave == 0) __builtin_amdgcn_s_wait_tensorcnt(0);
  }
  __syncthreads();
#else
  const int arow = threadIdx.x >> 1, aseg = (threadIdx.x & 1) * 16;
#endif

  for (int it = 0; it < NIT; ++it) {
    const int k0 = it * 32;
#if USE_TDM
    const int cur = it & 1;
    if (it + 1 < NIT) {                // pipeline next tiles during compute
      if (wave == 0)
        tdm_load_2d(Wo + (size_t)o0 * FEAT + k0 + 32,
                    (unsigned)(size_t)&at[cur ^ 1][0][0],
                    512, 1024, 512, 16, 128, 3, 3);
      union { uint4 u[2]; _Float16 h[16]; } tb;
      tb.u[0] = *(const uint4*)(Arow + k0 + 32);
      tb.u[1] = *(const uint4*)(Arow + k0 + 40);
      #pragma unroll
      for (int e = 0; e < 16; ++e) bt[cur ^ 1][bks + e][bww] = tb.h[e];
    }
#else
    const int cur = 0;
    __syncthreads();
    const _Float16* ga = Wo + (size_t)(o0 + arow) * FEAT + k0 + aseg;
    *(uint4*)&at[0][arow][aseg]     = *(const uint4*)ga;
    *(uint4*)&at[0][arow][aseg + 8] = *(const uint4*)(ga + 8);
    union { uint4 u[2]; _Float16 h[16]; } tb;
    tb.u[0] = *(const uint4*)(Arow + k0);
    tb.u[1] = *(const uint4*)(Arow + k0 + 8);
    #pragma unroll
    for (int e = 0; e < 16; ++e) bt[0][bks + e][bww] = tb.h[e];
    if (k0 + 32 < FEAT) __builtin_prefetch(Arow + k0 + 32, 0, 1);
    __syncthreads();
#endif
    v16h af;
    {
      int m = wave * 16 + (lane & 15);
      int hs = (lane >> 4) << 3;
      #pragma unroll
      for (int j = 0; j < 8; ++j) {
        int kbi = ((j & 4) << 2) + hs + ((j & 3) << 1);
        ((unsigned int*)&af)[j] = *(const unsigned int*)&at[cur][m][kbi];
      }
    }
    #pragma unroll
    for (int nt = 0; nt < 8; ++nt) {
      v16h bf;
      #pragma unroll
      for (int e = 0; e < 16; ++e)
        bf[e] = bt[cur][((e & 8) << 1) + ((lane >> 4) << 3) + (e & 7)][nt * 16 + (lane & 15)];
      cf[nt] = __builtin_amdgcn_wmma_f32_16x16x32_f16(false, af, false, bf,
                                                      (short)0, cf[nt], false, false);
    }
#if USE_TDM
    if (wave == 0) __builtin_amdgcn_s_wait_tensorcnt(0);
    __syncthreads();                    // retire readers + publish next buffer
#endif
  }
  const int mrow = o0 + wave * 16 + 8 * (lane >> 4);
  const int ncol = lane & 15;
  float* outb = Out + (size_t)b * FEAT * SEQL;
  #pragma unroll
  for (int nt = 0; nt < 8; ++nt)
    #pragma unroll
    for (int r = 0; r < 8; ++r)
      outb[(size_t)(mrow + r) * SEQL + w0 + nt * 16 + ncol] = cf[nt][r];
}

// ------------------------------------------------------------------- launcher
extern "C" void kernel_launch(void* const* d_in, const int* in_sizes, int n_in,
                              void* d_out, int out_size, void* d_ws, size_t ws_size,
                              hipStream_t stream) {
  const float* x   = (const float*)d_in[0];
  const float* qw  = (const float*)d_in[1];
  const float* kw  = (const float*)d_in[2];
  const float* vw  = (const float*)d_in[3];
  const float* vcw = (const float*)d_in[4];
  const float* ow  = (const float*)d_in[5];
  float* out = (float*)d_out;

  char* ws = (char*)d_ws;
  size_t off = 0;
  auto take = [&](size_t bytes) {
    char* p = ws + off;
    off = (off + bytes + 255) & ~(size_t)255;
    return p;
  };
  _Float16* xh   = (_Float16*)take((size_t)BATCH * FEAT * SEQL * 2);
  _Float16* vwh  = (_Float16*)take((size_t)FEAT * FEAT * 2);
  _Float16* owh  = (_Float16*)take((size_t)FEAT * FEAT * 2);
  _Float16* qbuf = (_Float16*)take((size_t)BATCH * FEAT * SEQL * 2);
  _Float16* kbuf = (_Float16*)take((size_t)BATCH * FEAT * SEQL * 2);
  _Float16* vbuf = (_Float16*)take((size_t)BATCH * FEAT * SEQL * 2);
  float*    v0   = (float*)   take((size_t)BATCH * FEAT * SEQL * 4);
  _Float16* abuf = (_Float16*)take((size_t)BATCH * FEAT * SEQL * 2);
  (void)ws_size; (void)n_in; (void)in_sizes; (void)out_size;

  const int nx = BATCH * FEAT * SEQL;
  const int nw = FEAT * FEAT;
  cast_f32_to_f16<<<nx / 1024, 256, 0, stream>>>(x, xh, nx);
  cast_f32_to_f16<<<nw / 1024, 256, 0, stream>>>(vw, vwh, nw);
  cast_f32_to_f16<<<nw / 1024, 256, 0, stream>>>(ow, owh, nw);

  const int nconv = BATCH * (FEAT / 2) * SEQL;
  qkconv_rope_kernel<<<nconv / 256, 256, 0, stream>>>(x, qw, kw, qbuf, kbuf);
  gemm_vproj_kernel<<<dim3(SEQL / 128, FEAT / 128, BATCH), 256, 0, stream>>>(vwh, xh, v0);
  vconv_kernel<<<nconv / 256, 256, 0, stream>>>(v0, vcw, vbuf);
  attn_kernel<<<dim3(SEQL / 128, NHEAD, BATCH), 256, 0, stream>>>(qbuf, kbuf, vbuf, abuf);
  gemm_outproj_kernel<<<dim3(SEQL / 128, FEAT / 128, BATCH), 256, 0, stream>>>(owh, abuf, out);
}
